// Atten2Map_28492813042210
// MI455X (gfx1250) — matrix operand order repaired and checked by reading, not compiled, not run on hardware
//
#include <hip/hip_runtime.h>
#include <hip/hip_bf16.h>

// Problem constants (from reference)
constexpr int kND   = 64;    // attention dim
constexpr int kNH   = 4;     // heads
constexpr int kNNEI = 128;   // neighbors (n, m)
constexpr int kDIN  = 64;    // g2 feature dim
constexpr float kSHIFT = 20.0f;

typedef __bf16 bf16x16 __attribute__((ext_vector_type(16)));
typedef __bf16 bf16x8  __attribute__((ext_vector_type(8)));
typedef float  f32x8   __attribute__((ext_vector_type(8)));
typedef unsigned int u32x4 __attribute__((ext_vector_type(4)));
typedef int i32x4 __attribute__((ext_vector_type(4)));
typedef int i32x8 __attribute__((ext_vector_type(8)));

// round-to-nearest-even f32 -> bf16 (bit pattern as ushort)
__device__ __forceinline__ unsigned short f2bf(float f) {
  unsigned u = __float_as_uint(f);
  u += 0x7FFFu + ((u >> 16) & 1u);
  return (unsigned short)(u >> 16);
}

union ABFrag { bf16x16 v; bf16x8 h[2]; };

// A-matrix 16x32 bf16 fragment: lane m = lane%16, g = lane/16.
// element e <-> K = k0 + g*16 + e (contiguous, matches B)
__device__ __forceinline__ bf16x16 load_a_frag(const unsigned short* rowptr, int k0, int g) {
  return *reinterpret_cast<const bf16x16*>(rowptr + k0 + g * 16);
}

// B-matrix 32x16 bf16 fragment: lane n = lane%16, g = lane/16,
// element e <-> K = k0 + g*16 + e (contiguous). rowptr = &Bt[n][0] ([N][K] layout).
__device__ __forceinline__ bf16x16 load_b_frag(const unsigned short* rowptr, int k0, int g) {
  return *reinterpret_cast<const bf16x16*>(rowptr + k0 + g * 16);
}

__device__ __forceinline__ f32x8 wmma_bf16(bf16x16 a, bf16x16 b, f32x8 c) {
  // (neg_a, A, neg_b, B, c_mod, C, reuse_a, reuse_b)
  return __builtin_amdgcn_wmma_f32_16x16x32_bf16(false, a, false, b, (short)0, c, false, false);
}

__global__ void __launch_bounds__(256, 1)
atten2map_kernel(const float* __restrict__ g2,
                 const float* __restrict__ h2,
                 const int*   __restrict__ nmask,
                 const float* __restrict__ sw,
                 const float* __restrict__ wqk,
                 float*       __restrict__ out)
{
  __shared__ float          sStage[kNNEI][kDIN];   // TDM landing zone (g2 f32), 32 KB
  __shared__ unsigned short sG2[kNNEI][kDIN];      // g2 tile, bf16, 16 KB
  __shared__ unsigned short sWt[2 * kND][kDIN];    // W_h transposed [c][d], bf16, 16 KB
  __shared__ unsigned short sQ[kNNEI][kND];        // Q_h rows, bf16, 16 KB
  __shared__ unsigned short sK[kNNEI][kND];        // K_h rows, bf16, 16 KB
  __shared__ float sH2[kNNEI][4];                  // h2 rows (3 used)
  __shared__ float sSw[kNNEI];
  __shared__ int   sMask[kNNEI];

  const int bl   = blockIdx.x;          // fused (b, l)
  const int tid  = threadIdx.x;
  const int lane = tid & 31;
  const int w    = tid >> 5;            // wave id, owns rows w*16 .. w*16+15
  const int g    = lane >> 4;           // half-wave
  const int ln   = lane & 15;

  const float* g2p  = g2 + (size_t)bl * kNNEI * kDIN;
  const float* h2p  = h2 + (size_t)bl * kNNEI * 3;
  const int*   mp   = nmask + (size_t)bl * kNNEI;
  const float* swp  = sw + (size_t)bl * kNNEI;
  float*       outp = out + (size_t)bl * kNNEI * kNNEI * kNH;

  // Warm Wqk (128 KB, shared by all 1024 blocks) into L2: global_prefetch_b8
  #pragma unroll
  for (int i = 0; i < 4; ++i)
    __builtin_prefetch(wqk + ((size_t)i * 256 + tid) * 32, 0, 3);

  // ---- Async-tensor load of the g2 tile (128x64 f32, contiguous) into LDS.
  // 2D Tensor DMA descriptor (ISA ch.8): issued once, by wave 0 (EXEC ignored).
  if (w == 0) {
    const unsigned lds_off = (unsigned)(uintptr_t)(&sStage[0][0]); // flat LDS addr low 32b = WG offset
    const unsigned long long ga = (unsigned long long)(uintptr_t)g2p;
    u32x4 d0;
    d0[0] = 1u;                                         // count=1 (valid), user mode
    d0[1] = lds_off;                                    // lds_addr (bytes)
    d0[2] = (unsigned)(ga & 0xFFFFFFFFu);               // global_addr[31:0]
    d0[3] = (unsigned)((ga >> 32) & 0x01FFFFFFu)        // global_addr[56:32]
          | 0x80000000u;                                // type = 2 (bits 127:126 = 10b)
    i32x8 d1;
    d1[0] = 0x00020000;                                 // workgroup_mask=0, data_size=2 (4B)
    d1[1] = (int)(64u << 16);                           // tensor_dim0 = 64 elements/row
    d1[2] = (int)(128u << 16);                          // tensor_dim1 = 128 rows
    d1[3] = (int)(64u << 16);                           // tile_dim0 = 64
    d1[4] = 128;                                        // tile_dim1 = 128, tile_dim2 = 0
    d1[5] = 64;                                         // tensor_dim0_stride = 64
    d1[6] = 0;
    d1[7] = 0;
    const i32x4 zz4 = {0, 0, 0, 0};                     // groups 2/3 unused (2D)
    const i32x8 zz8 = {0, 0, 0, 0, 0, 0, 0, 0};
    __builtin_amdgcn_tensor_load_to_lds(d0, d1, zz4, zz4, zz8, 0);
  }

  // Overlap with the DMA: stage h2 / sw / mask through the normal path
  for (int i = tid; i < kNNEI * 3; i += 256)
    sH2[i / 3][i % 3] = h2p[i];
  if (tid < kNNEI) { sSw[tid] = swp[tid]; sMask[tid] = mp[tid]; }

  if (w == 0) __builtin_amdgcn_s_wait_tensorcnt(0);
  __syncthreads();

  // Convert staged f32 g2 tile -> bf16 for WMMA A-fragments
  for (int i = tid; i < kNNEI * kDIN; i += 256)
    (&sG2[0][0])[i] = f2bf((&sStage[0][0])[i]);
  __syncthreads();

  // Per-lane row-side constants (rows rowbase+0..7 for this lane's half-wave)
  const int rowbase = w * 16 + 8 * g;
  float h2n0[8], h2n1[8], h2n2[8], swn[8];
  int mkn[8];
  #pragma unroll
  for (int r = 0; r < 8; ++r) {
    const int n = rowbase + r;
    h2n0[r] = sH2[n][0]; h2n1[r] = sH2[n][1]; h2n2[r] = sH2[n][2];
    swn[r]  = sSw[n];    mkn[r]  = sMask[n];
  }

  for (int h = 0; h < kNH; ++h) {
    // ---- Gather this head's W columns, transposed: sWt[c][d] = Wqk[d][col(c,h)]
    // c < 64 : q-dim c    -> Wqk column c*8 + h
    // c >= 64: k-dim c-64 -> Wqk column (c-64)*8 + 4 + h
    for (int i = tid; i < 2 * kND * kDIN; i += 256) {
      const int c = i >> 6, d = i & 63;
      const int col = (c < kND) ? (c * 2 * kNH + h) : ((c - kND) * 2 * kNH + kNH + h);
      sWt[c][d] = f2bf(wqk[d * (kND * 2 * kNH) + col]);
    }
    __syncthreads();

    // ---- GEMM1: [Q|K] = g2 @ W_h  (each wave: its 16-row strip x 128 cols)
    {
      const unsigned short* arow = &sG2[w * 16 + ln][0];
      #pragma unroll
      for (int nt = 0; nt < 8; ++nt) {
        f32x8 c1 = {0.f, 0.f, 0.f, 0.f, 0.f, 0.f, 0.f, 0.f};
        bf16x16 a0 = load_a_frag(arow, 0, g);
        bf16x16 b0 = load_b_frag(&sWt[nt * 16 + ln][0], 0, g);
        c1 = wmma_bf16(a0, b0, c1);
        bf16x16 a1 = load_a_frag(arow, 32, g);
        bf16x16 b1 = load_b_frag(&sWt[nt * 16 + ln][0], 32, g);
        c1 = wmma_bf16(a1, b1, c1);
        const int cc = nt * 16 + ln;
        #pragma unroll
        for (int r = 0; r < 8; ++r) {
          const int n = rowbase + r;            // C layout: row = r + 8*(lane/16)
          const unsigned short v = f2bf(c1[r]);
          if (nt < 4) sQ[n][cc] = v; else sK[n][cc - kND] = v;
        }
      }
    }
    __syncthreads();

    // ---- GEMM2: S = Q @ K^T (K stored [m][d] => contiguous B-fragments)
    f32x8 acc[8];
    #pragma unroll
    for (int j = 0; j < 8; ++j) acc[j] = (f32x8){0.f, 0.f, 0.f, 0.f, 0.f, 0.f, 0.f, 0.f};
    {
      const unsigned short* qrow = &sQ[w * 16 + ln][0];
      bf16x16 qa0 = load_a_frag(qrow, 0, g);
      bf16x16 qa1 = load_a_frag(qrow, 32, g);
      #pragma unroll
      for (int j = 0; j < 8; ++j) {
        bf16x16 kb0 = load_b_frag(&sK[j * 16 + ln][0], 0, g);
        bf16x16 kb1 = load_b_frag(&sK[j * 16 + ln][0], 32, g);
        acc[j] = wmma_bf16(qa0, kb0, acc[j]);
        acc[j] = wmma_bf16(qa1, kb1, acc[j]);
      }
    }

    // ---- Epilogue: logits, softmax over m, mask, output
    float rmax[8];
    #pragma unroll
    for (int r = 0; r < 8; ++r) rmax[r] = -3.0e38f;

    #pragma unroll
    for (int j = 0; j < 8; ++j) {
      const int m = j * 16 + ln;
      const float m0 = sH2[m][0], m1 = sH2[m][1], m2 = sH2[m][2];
      const float swm = sSw[m];
      #pragma unroll
      for (int r = 0; r < 8; ++r) {
        const float hh = h2n0[r] * m0 + h2n1[r] * m1 + h2n2[r] * m2;
        float s = acc[j][r] * 0.125f * hh;             // /sqrt(ND) then *hh
        s = (s + kSHIFT) * swn[r] * swm - kSHIFT;
        acc[j][r] = s;
        rmax[r] = fmaxf(rmax[r], s);
      }
    }
    #pragma unroll
    for (int d = 1; d < 16; d <<= 1)
      #pragma unroll
      for (int r = 0; r < 8; ++r)
        rmax[r] = fmaxf(rmax[r], __shfl_xor(rmax[r], d, 16));

    float rsum[8];
    #pragma unroll
    for (int r = 0; r < 8; ++r) rsum[r] = 0.f;
    #pragma unroll
    for (int j = 0; j < 8; ++j)
      #pragma unroll
      for (int r = 0; r < 8; ++r) {
        const float e = __expf(acc[j][r] - rmax[r]);
        acc[j][r] = e;
        rsum[r] += e;
      }
    #pragma unroll
    for (int d = 1; d < 16; d <<= 1)
      #pragma unroll
      for (int r = 0; r < 8; ++r)
        rsum[r] += __shfl_xor(rsum[r], d, 16);
    float rinv[8];
    #pragma unroll
    for (int r = 0; r < 8; ++r) rinv[r] = 1.f / rsum[r];

    #pragma unroll
    for (int j = 0; j < 8; ++j) {
      const int m = j * 16 + ln;
      const float m0 = sH2[m][0], m1 = sH2[m][1], m2 = sH2[m][2];
      const float swm = sSw[m];
      const int mkm = sMask[m];
      #pragma unroll
      for (int r = 0; r < 8; ++r) {
        const int n = rowbase + r;
        const float hh = h2n0[r] * m0 + h2n1[r] * m1 + h2n2[r] * m2;
        float p = acc[j][r] * rinv[r];
        p = (mkm && mkn[r]) ? (p * swn[r] * swm) : 0.f;
        outp[((size_t)(n * kNNEI + m)) * kNH + h] = p * hh * 0.57735026918962576f; // hh/sqrt(3)
      }
    }
    __syncthreads();  // protect sWt/sQ/sK for next head
  }
}

extern "C" void kernel_launch(void* const* d_in, const int* in_sizes, int n_in,
                              void* d_out, int out_size, void* d_ws, size_t ws_size,
                              hipStream_t stream) {
  const float* g2    = (const float*)d_in[0];
  const float* h2    = (const float*)d_in[1];
  const int*   nmask = (const int*)d_in[2];
  const float* sw    = (const float*)d_in[3];
  const float* wqk   = (const float*)d_in[4];
  float* out = (float*)d_out;

  const int nbl = in_sizes[0] / (kNNEI * kDIN);   // nb*nloc = 1024
  atten2map_kernel<<<nbl, 256, 0, stream>>>(g2, h2, nmask, sw, wqk, out);
}
